// GATv2_2473901163013
// MI455X (gfx1250) — matrix operand order repaired and checked by reading, not compiled, not run on hardware
//
#include <hip/hip_runtime.h>
#include <hip/hip_bf16.h>

// ---------------------------------------------------------------------------
// CDNA5 (gfx1250) GAT forward: bf16 WMMA GEMMs, LDS-staged weights,
// fused attention epilogue.
// ---------------------------------------------------------------------------

typedef __bf16 v16bf __attribute__((ext_vector_type(16)));
typedef float  v8f   __attribute__((ext_vector_type(8)));

union BFrag { unsigned short u[16]; v16bf v; };

#define WPITCH 136   // LDS row pitch in ushorts (272 B = 68 dwords -> no bank conflicts)

// f32 -> bf16, round-to-nearest-even
__device__ __forceinline__ unsigned short f2bf(float x) {
    unsigned b = __float_as_uint(x);
    return (unsigned short)((b + 0x7FFFu + ((b >> 16) & 1u)) >> 16);
}

// monotonic float <-> uint key for atomic max on floats
__device__ __forceinline__ unsigned f2key(float x) {
    unsigned b = __float_as_uint(x);
    return (b & 0x80000000u) ? ~b : (b | 0x80000000u);
}
__device__ __forceinline__ float key2f(unsigned k) {
    unsigned b = (k & 0x80000000u) ? (k & 0x7FFFFFFFu) : ~k;
    return __uint_as_float(b);
}

// ---------------------------------------------------------------------------
// Kernel 0: convert 128x128 f32 weights -> transposed bf16 (WT[n][k], linear)
// ---------------------------------------------------------------------------
__global__ void __launch_bounds__(256)
conv_weights(const float* __restrict__ Ws, const float* __restrict__ Wr,
             const float* __restrict__ We,
             unsigned short* __restrict__ WsT, unsigned short* __restrict__ WrT,
             unsigned short* __restrict__ WeT) {
    int tid = blockIdx.x * blockDim.x + threadIdx.x;
    if (tid >= 128 * 128) return;
    int n = tid >> 7, k = tid & 127;
    WsT[tid] = f2bf(Ws[k * 128 + n]);
    WrT[tid] = f2bf(Wr[k * 128 + n]);
    WeT[tid] = f2bf(We[k * 128 + n]);
}

// ---------------------------------------------------------------------------
// Kernel 1: zero-init out [N*128], denom [N*8], mxkey [N*8]
// ---------------------------------------------------------------------------
__global__ void __launch_bounds__(256)
init_zero(unsigned* __restrict__ mxkey, float* __restrict__ denom,
          float* __restrict__ out, int N) {
    int tid = blockIdx.x * blockDim.x + threadIdx.x;
    if (tid < N * 128) out[tid] = 0.0f;
    if (tid < N * 8) { mxkey[tid] = 0u; denom[tid] = 0.0f; }
}

// ---------------------------------------------------------------------------
// Load 16-bit A fragment (16x32 tile, bf16) from f32 row-major source.
// Lane L: row = (L&15); K = (L>>4)*16..+15 (contiguous).
// ---------------------------------------------------------------------------
__device__ __forceinline__ void load_a_frag(BFrag& A, const float* __restrict__ row,
                                            int ktile, int hilo) {
    const float* p = row + ktile * 32 + hilo * 16;
#pragma unroll
    for (int j = 0; j < 8; ++j) {
        A.u[j]     = f2bf(p[j]);
        A.u[8 + j] = f2bf(p[8 + j]);
    }
}

// ---------------------------------------------------------------------------
// Kernel 2: node projections  hs = nodes@Ws + bs,  hr = nodes@Wr + br
// Weights staged in LDS (pitch-padded); one wave per 16-row tile; A fragments
// shared across both weight matrices.
// ---------------------------------------------------------------------------
__global__ void __launch_bounds__(256)
node_proj(const float* __restrict__ nodes,
          const unsigned short* __restrict__ WsT, const unsigned short* __restrict__ WrT,
          const float* __restrict__ Ws_b, const float* __restrict__ Wr_b,
          float* __restrict__ hs, float* __restrict__ hr, int N) {
    __shared__ unsigned short sWs[128 * WPITCH];
    __shared__ unsigned short sWr[128 * WPITCH];

    // Stage both weight matrices: 2048 uint4 (16 B) chunks each.
    {
        const uint4* gs = (const uint4*)WsT;
        const uint4* gr = (const uint4*)WrT;
        for (int i = threadIdx.x; i < 2048; i += 256) {
            int n = i >> 4, kq = (i & 15) * 8;
            *(uint4*)&sWs[n * WPITCH + kq] = gs[i];
            *(uint4*)&sWr[n * WPITCH + kq] = gr[i];
        }
    }
    __syncthreads();

    int wave = blockIdx.x * (blockDim.x >> 5) + (threadIdx.x >> 5);
    int row0 = wave * 16;
    if (row0 >= N) return;                    // wave-uniform, after the only barrier
    int lane = threadIdx.x & 31;
    int m = lane & 15, hilo = lane >> 4;

    int rowA = row0 + m; if (rowA >= N) rowA = N - 1;
    const float* arow = nodes + (size_t)rowA * 128;

    BFrag A[4];
#pragma unroll
    for (int kt = 0; kt < 4; ++kt) load_a_frag(A[kt], arow, kt, hilo);

#pragma unroll
    for (int nt = 0; nt < 8; ++nt) {
        v8f cs = {};
        v8f cr = {};
        int ncol = nt * 16 + m;
#pragma unroll
        for (int kt = 0; kt < 4; ++kt) {
            BFrag Bs, Br;
            int boff = ncol * WPITCH + kt * 32 + hilo * 16;
#pragma unroll
            for (int j = 0; j < 16; ++j) { Bs.u[j] = sWs[boff + j]; Br.u[j] = sWr[boff + j]; }
            cs = __builtin_amdgcn_wmma_f32_16x16x32_bf16(false, A[kt].v, false, Bs.v,
                                                         (short)0, cs, false, false);
            cr = __builtin_amdgcn_wmma_f32_16x16x32_bf16(false, A[kt].v, false, Br.v,
                                                         (short)0, cr, false, false);
        }
        float bs = Ws_b[ncol], br = Wr_b[ncol];
#pragma unroll
        for (int r = 0; r < 8; ++r) {
            int row = row0 + r + 8 * hilo;
            if (row < N) {
                hs[(size_t)row * 128 + ncol] = cs[r] + bs;
                hr[(size_t)row * 128 + ncol] = cr[r] + br;
            }
        }
    }
}

// ---------------------------------------------------------------------------
// Kernel 3: edge pass. he = edges@We + be (WMMA, We in LDS), fused with
// x = leaky_relu(he + hs[snd] + hr[rcv]); logit[h] = <x_head_h, a_h>.
// N-tile nt == head h (HEAD_DIM == 16). Segment-max via atomic u32 keys.
// ---------------------------------------------------------------------------
__global__ void __launch_bounds__(256)
edge_pass(const float* __restrict__ edges, const unsigned short* __restrict__ WeT,
          const float* __restrict__ We_b,
          const float* __restrict__ hs, const float* __restrict__ hr,
          const int* __restrict__ senders, const int* __restrict__ receivers,
          const float* __restrict__ a,          // [8][16]
          float* __restrict__ logits,           // [E][8]
          unsigned* __restrict__ mxkey,         // [N][8]
          int E) {
    __shared__ unsigned short sWe[128 * WPITCH];
    {
        const uint4* g = (const uint4*)WeT;
        for (int i = threadIdx.x; i < 2048; i += 256) {
            int n = i >> 4, kq = (i & 15) * 8;
            *(uint4*)&sWe[n * WPITCH + kq] = g[i];
        }
    }
    __syncthreads();

    int wave = blockIdx.x * (blockDim.x >> 5) + (threadIdx.x >> 5);
    int e0 = wave * 16;
    if (e0 >= E) return;                      // wave-uniform, after the only barrier
    int lane = threadIdx.x & 31;
    int m = lane & 15, hilo = lane >> 4;

    int erow = e0 + m; if (erow >= E) erow = E - 1;
    const float* arow = edges + (size_t)erow * 128;

    BFrag A[4];
#pragma unroll
    for (int kt = 0; kt < 4; ++kt) load_a_frag(A[kt], arow, kt, hilo);

    int snd[8], rcv[8];
#pragma unroll
    for (int r = 0; r < 8; ++r) {
        int e = e0 + r + 8 * hilo; if (e >= E) e = E - 1;
        snd[r] = senders[e];
        rcv[r] = receivers[e];
    }

#pragma unroll
    for (int h = 0; h < 8; ++h) {
        v8f c = {};
        int ncol = h * 16 + m;                // embed column == (head h, head-dim m)
#pragma unroll
        for (int kt = 0; kt < 4; ++kt) {
            BFrag B;
            int boff = ncol * WPITCH + kt * 32 + hilo * 16;
#pragma unroll
            for (int j = 0; j < 16; ++j) B.u[j] = sWe[boff + j];
            c = __builtin_amdgcn_wmma_f32_16x16x32_bf16(false, A[kt].v, false, B.v,
                                                        (short)0, c, false, false);
        }
        float bias = We_b[ncol];
        float ah = a[ncol];                   // a[h][m]
#pragma unroll
        for (int r = 0; r < 8; ++r) {
            float x = c[r] + bias
                    + hs[(size_t)snd[r] * 128 + ncol]
                    + hr[(size_t)rcv[r] * 128 + ncol];
            x = (x > 0.0f) ? x : 0.01f * x;   // leaky_relu slope 0.01
            float contrib = x * ah;
            // reduce over the 16 lanes of this row group (head dim)
            contrib += __shfl_xor(contrib, 1, 32);
            contrib += __shfl_xor(contrib, 2, 32);
            contrib += __shfl_xor(contrib, 4, 32);
            contrib += __shfl_xor(contrib, 8, 32);
            if (m == 0) {
                int e = e0 + r + 8 * hilo;
                if (e < E) {
                    logits[(size_t)e * 8 + h] = contrib;
                    atomicMax(&mxkey[(size_t)rcv[r] * 8 + h], f2key(contrib));
                }
            }
        }
    }
}

// ---------------------------------------------------------------------------
// Kernel 4: ex = exp(logit - mx[rcv]); denom[rcv] += ex; out[rcv] += ex*sent
// (unnormalized accumulation; normalized in kernel 5). One thread = (edge,head).
// ---------------------------------------------------------------------------
__global__ void __launch_bounds__(256)
softmax_accum(const float* __restrict__ logits, const unsigned* __restrict__ mxkey,
              const int* __restrict__ senders, const int* __restrict__ receivers,
              const float* __restrict__ hs,
              float* __restrict__ denom, float* __restrict__ out, int E) {
    int tid = blockIdx.x * blockDim.x + threadIdx.x;
    if (tid >= E * 8) return;
    int e = tid >> 3, h = tid & 7;
    int rcv = receivers[e], snd = senders[e];
    unsigned k = mxkey[(size_t)rcv * 8 + h];
    float mx = k ? key2f(k) : 0.0f;           // untouched -> 0 (matches reference)
    float ex = __expf(logits[tid] - mx);
    atomicAdd(&denom[(size_t)rcv * 8 + h], ex);
    const float* s = hs + (size_t)snd * 128 + h * 16;
    float* o = out + (size_t)rcv * 128 + h * 16;
#pragma unroll
    for (int d = 0; d < 16; ++d) atomicAdd(o + d, ex * s[d]);
}

// ---------------------------------------------------------------------------
// Kernel 5: out /= denom (per node, per head); empty segments stay 0.
// ---------------------------------------------------------------------------
__global__ void __launch_bounds__(256)
normalize(float* __restrict__ out, const float* __restrict__ denom, int N) {
    int tid = blockIdx.x * blockDim.x + threadIdx.x;
    if (tid >= N * 128) return;
    int n = tid >> 7, h = (tid >> 4) & 7;
    float dn = denom[(size_t)n * 8 + h];
    float v = out[tid];
    out[tid] = (dn > 0.0f) ? v / dn : 0.0f;
}

// ---------------------------------------------------------------------------
extern "C" void kernel_launch(void* const* d_in, const int* in_sizes, int n_in,
                              void* d_out, int out_size, void* d_ws, size_t ws_size,
                              hipStream_t stream) {
    const float* nodes     = (const float*)d_in[0];
    const float* edges     = (const float*)d_in[1];
    const int*   senders   = (const int*)d_in[2];
    const int*   receivers = (const int*)d_in[3];
    const float* Ws_k      = (const float*)d_in[4];
    const float* Ws_b      = (const float*)d_in[5];
    const float* Wr_k      = (const float*)d_in[6];
    const float* Wr_b      = (const float*)d_in[7];
    const float* We_k      = (const float*)d_in[8];
    const float* We_b      = (const float*)d_in[9];
    const float* a         = (const float*)d_in[10];
    float* out = (float*)d_out;

    const int N = in_sizes[0] / 128;
    const int E = in_sizes[1] / 128;

    char* ws = (char*)d_ws;
    float* hs            = (float*)ws;          ws += (size_t)N * 128 * sizeof(float);
    float* hr            = (float*)ws;          ws += (size_t)N * 128 * sizeof(float);
    float* logits        = (float*)ws;          ws += (size_t)E * 8   * sizeof(float);
    float* denom         = (float*)ws;          ws += (size_t)N * 8   * sizeof(float);
    unsigned* mxkey      = (unsigned*)ws;       ws += (size_t)N * 8   * sizeof(unsigned);
    unsigned short* WsT  = (unsigned short*)ws; ws += 128 * 128 * sizeof(unsigned short);
    unsigned short* WrT  = (unsigned short*)ws; ws += 128 * 128 * sizeof(unsigned short);
    unsigned short* WeT  = (unsigned short*)ws;

    conv_weights<<<(128 * 128 + 255) / 256, 256, 0, stream>>>(Ws_k, Wr_k, We_k,
                                                              WsT, WrT, WeT);
    init_zero<<<(N * 128 + 255) / 256, 256, 0, stream>>>(mxkey, denom, out, N);

    int nodeWaves = (N + 15) / 16;
    node_proj<<<(nodeWaves + 7) / 8, 256, 0, stream>>>(nodes, WsT, WrT, Ws_b, Wr_b,
                                                       hs, hr, N);
    int edgeWaves = (E + 15) / 16;
    edge_pass<<<(edgeWaves + 7) / 8, 256, 0, stream>>>(edges, WeT, We_b, hs, hr,
                                                       senders, receivers, a,
                                                       logits, mxkey, E);
    softmax_accum<<<(E * 8 + 255) / 256, 256, 0, stream>>>(logits, mxkey, senders,
                                                           receivers, hs, denom, out, E);
    normalize<<<(N * 128 + 255) / 256, 256, 0, stream>>>(out, denom, N);
}
